// semiMPSN_29257317220560
// MI455X (gfx1250) — matrix operand order repaired and checked by reading, not compiled, not run on hardware
//
#include <hip/hip_runtime.h>
#include <hip/hip_bf16.h>

// ---------------------------------------------------------------------------
// CDNA5 (gfx1250, wave32) simplicial message-passing net.
//  - Dense [N,128]x[128,128] via v_wmma_f32_16x16x32_bf16, weights staged in
//    LDS (fragment-contiguous, conflict-free), activations bf16 in memory.
//  - COO SpMM: one wave per nnz, bf16 gather + f32 global_atomic_add scatter.
// ---------------------------------------------------------------------------

typedef __attribute__((ext_vector_type(16))) __bf16 v16bf;
typedef __attribute__((ext_vector_type(8)))  float  v8f;

#define DIM 128

__device__ __forceinline__ unsigned short f2bf(float f) {
  unsigned u = __builtin_bit_cast(unsigned, f);
  u += 0x7FFFu + ((u >> 16) & 1u);        // round-to-nearest-even
  return (unsigned short)(u >> 16);
}
__device__ __forceinline__ float bf2f_lo(unsigned u) {
  return __builtin_bit_cast(float, u << 16);
}
__device__ __forceinline__ float bf2f_hi(unsigned u) {
  return __builtin_bit_cast(float, u & 0xFFFF0000u);
}

// ------------------------------- zero fill ---------------------------------
__global__ void zero_kernel(float4* p, long long n4) {
  long long i = (long long)blockIdx.x * blockDim.x + threadIdx.x;
  long long stride = (long long)gridDim.x * blockDim.x;
  float4 z = make_float4(0.f, 0.f, 0.f, 0.f);
  for (; i < n4; i += stride) p[i] = z;
}

// ---------------- weight prep: f32 -> bf16, and Whead = Wfc @ Wcls ----------
__global__ void prep_w_kernel(const float* __restrict__ W,
                              const float* __restrict__ Wfc,
                              const float* __restrict__ Wcls,
                              unsigned short* __restrict__ Wbf,
                              unsigned short* __restrict__ Whead, int nc) {
  int b = blockIdx.x;
  if (b < 54) {  // 3 layers * 3 dims * 6 terms
    const float* src = W + (size_t)b * DIM * DIM;
    unsigned short* dst = Wbf + (size_t)b * DIM * DIM;
    for (int i = threadIdx.x; i < DIM * DIM; i += blockDim.x) dst[i] = f2bf(src[i]);
  } else {       // Whead[k][n] = sum_j Wfc[k][j] * Wcls[j][n]   (128 x nc)
    for (int i = threadIdx.x; i < DIM * nc; i += blockDim.x) {
      int k = i / nc, n = i % nc;
      float s = 0.f;
      for (int j = 0; j < DIM; ++j) s += Wfc[k * DIM + j] * Wcls[j * nc + n];
      Whead[i] = f2bf(s);
    }
  }
}

// -------- f32 -> bf16 row conversion; optional fused 1/max(cnt,1) scale ----
__global__ void cvt_bf16_kernel(const float* __restrict__ src,
                                unsigned short* __restrict__ dst,
                                const float* __restrict__ cnt, long long n4) {
  long long i = (long long)blockIdx.x * blockDim.x + threadIdx.x;
  long long stride = (long long)gridDim.x * blockDim.x;
  for (; i < n4; i += stride) {
    float4 f = ((const float4*)src)[i];
    if (cnt) {
      float s = 1.0f / fmaxf(cnt[(i * 4) / DIM], 1.0f);
      f.x *= s; f.y *= s; f.z *= s; f.w *= s;
    }
    uint2 o;
    o.x = (unsigned)f2bf(f.x) | ((unsigned)f2bf(f.y) << 16);
    o.y = (unsigned)f2bf(f.z) | ((unsigned)f2bf(f.w) << 16);
    ((uint2*)dst)[i] = o;
  }
}

// ------------------------------- COO SpMM ----------------------------------
// out[r,:] += v * x[c,:] ; one wave32 per nnz, 4 cols/lane; x bf16, out f32.
__global__ void spmm_kernel(const int* __restrict__ rows, const int* __restrict__ cols,
                            const float* __restrict__ vals,
                            const unsigned short* __restrict__ x,
                            float* __restrict__ out, int nnz) {
  int e = (int)((blockIdx.x * blockDim.x + threadIdx.x) >> 5);
  if (e >= nnz) return;
  int lane = threadIdx.x & 31;
  int r = rows[e], c = cols[e];
  float v = vals[e];
  uint2 xq = *(const uint2*)(x + (size_t)c * DIM + lane * 4);
  float* o = out + (size_t)r * DIM + lane * 4;
  atomicAdd(o + 0, v * bf2f_lo(xq.x));
  atomicAdd(o + 1, v * bf2f_hi(xq.x));
  atomicAdd(o + 2, v * bf2f_lo(xq.y));
  atomicAdd(o + 3, v * bf2f_hi(xq.y));
}

// --------------------------- hyperedge pooling -----------------------------
__global__ void pool_kernel(const int* __restrict__ he_idx, const int* __restrict__ he_seg,
                            const unsigned short* __restrict__ x0,
                            float* __restrict__ pooled, float* __restrict__ cnt, int tot) {
  int e = (int)((blockIdx.x * blockDim.x + threadIdx.x) >> 5);
  if (e >= tot) return;
  int lane = threadIdx.x & 31;
  int s = he_seg[e], i = he_idx[e];
  uint2 xq = *(const uint2*)(x0 + (size_t)i * DIM + lane * 4);
  float* o = pooled + (size_t)s * DIM + lane * 4;
  atomicAdd(o + 0, bf2f_lo(xq.x));
  atomicAdd(o + 1, bf2f_hi(xq.x));
  atomicAdd(o + 2, bf2f_lo(xq.y));
  atomicAdd(o + 3, bf2f_hi(xq.y));
  if (lane == 0) atomicAdd(cnt + s, 1.0f);
}

// ------------------------------ WMMA GEMM ----------------------------------
// out[M, NTILES*16] = act( sum_t A_t[M,128] @ B_t[128, NTILES*16] )
// A: bf16 row-major. B: bf16 row-major [K][N], staged per-term into LDS in
// fragment-contiguous order. Each wave: 32 rows x NTILES*16 cols.
struct GemmArgs {
  const unsigned short* A[6];
  const unsigned short* B[6];
  const float* bias;        // optional per-col bias (null = 0)
  void* out;
  int nterms;
  int M;
  int apply_tanh;
  int out_bf16;             // 1: write bf16, 0: write f32
  int drop_first;           // skip row 0, shift output up by one row (f32 out)
};

template <int NTILES>
__global__ __launch_bounds__(256) void gemm_wmma_kernel(GemmArgs g) {
  const int NOUT = NTILES * 16;
  __shared__ uint4 smem[NTILES * 256];     // NTILES*4KB, fragment-contiguous

  int tid = threadIdx.x;
  int lane = tid & 31;
  int wave = (int)(((size_t)blockIdx.x * blockDim.x + tid) >> 5);
  int m0 = wave * 32;
  bool active = (m0 < g.M);                // wave-uniform

  int khalf = (lane >> 4) << 3;            // A: K sub-offset 0 or 8
  int arow = m0 + (lane & 15);

  v8f acc[2][NTILES];
  v8f z8 = {0.f, 0.f, 0.f, 0.f, 0.f, 0.f, 0.f, 0.f};
#pragma unroll
  for (int mi = 0; mi < 2; ++mi)
#pragma unroll
    for (int nt = 0; nt < NTILES; ++nt) acc[mi][nt] = z8;

  for (int t = 0; t < g.nterms; ++t) {
    // ---- cooperative stage: B[K=128][NOUT] -> fragment-contiguous LDS ----
    // dest unit u (16B): L=u&31, h=(u>>5)&1, gg=u>>6, nt=gg%NTILES, kc=gg/NTILES
    // holds B col nt*16 + (L&15), rows k = kc*32 + 16h + 8*(L>>4) .. +7
    const unsigned short* bsrc = g.B[t];
    for (int u = tid; u < NTILES * 256; u += 256) {
      int L = u & 31, h = (u >> 5) & 1, gg = u >> 6;
      int ntb = gg % NTILES, kc = gg / NTILES;
      int k = kc * 32 + (h << 4) + ((L >> 4) << 3);
      const unsigned short* bp = bsrc + (size_t)k * NOUT + ntb * 16 + (L & 15);
      uint4 q;
      q.x = (unsigned)bp[0 * NOUT] | ((unsigned)bp[1 * NOUT] << 16);
      q.y = (unsigned)bp[2 * NOUT] | ((unsigned)bp[3 * NOUT] << 16);
      q.z = (unsigned)bp[4 * NOUT] | ((unsigned)bp[5 * NOUT] << 16);
      q.w = (unsigned)bp[6 * NOUT] | ((unsigned)bp[7 * NOUT] << 16);
      smem[u] = q;
    }
    __syncthreads();

    if (active) {
      const unsigned short* A = g.A[t];
#pragma unroll
      for (int kc = 0; kc < 4; ++kc) {
        union { v16bf v; uint4 q[2]; } af[2];
#pragma unroll
        for (int mi = 0; mi < 2; ++mi) {
          const unsigned short* ap =
              A + (size_t)(arow + mi * 16) * DIM + kc * 32 + khalf;
          af[mi].q[0] = *(const uint4*)(ap);
          af[mi].q[1] = *(const uint4*)(ap + 16);
        }
#pragma unroll
        for (int nt = 0; nt < NTILES; ++nt) {
          union { v16bf v; uint4 q[2]; } bfr;
          int gg = kc * NTILES + nt;
          bfr.q[0] = smem[gg * 64 + lane];
          bfr.q[1] = smem[gg * 64 + 32 + lane];
#pragma unroll
          for (int mi = 0; mi < 2; ++mi)
            acc[mi][nt] = __builtin_amdgcn_wmma_f32_16x16x32_bf16(
                false, af[mi].v, false, bfr.v, (short)0, acc[mi][nt], false, false);
        }
      }
    }
    __syncthreads();  // LDS reused by next term
  }

  if (active) {
    // C/D: VGPR r, lanes 0-15 -> M=r, lanes 16-31 -> M=r+8; N = lane&15
    int ncol = lane & 15;
    int rofs = (lane >> 4) << 3;
#pragma unroll
    for (int mi = 0; mi < 2; ++mi) {
#pragma unroll
      for (int nt = 0; nt < NTILES; ++nt) {
#pragma unroll
        for (int r = 0; r < 8; ++r) {
          float v = acc[mi][nt][r];
          if (g.apply_tanh) v = tanhf(v);
          int n = nt * 16 + ncol;
          if (g.bias) v += g.bias[n];
          int m = m0 + mi * 16 + rofs + r;
          if (g.out_bf16) {
            ((unsigned short*)g.out)[(size_t)m * NOUT + n] = f2bf(v);
          } else if (g.drop_first) {
            if (m > 0) ((float*)g.out)[(size_t)(m - 1) * NOUT + n] = v;
          } else {
            ((float*)g.out)[(size_t)m * NOUT + n] = v;
          }
        }
      }
    }
  }
}

// ---------------------------------------------------------------------------
extern "C" void kernel_launch(void* const* d_in, const int* in_sizes, int n_in,
                              void* d_out, int out_size, void* d_ws, size_t ws_size,
                              hipStream_t stream) {
  (void)n_in; (void)out_size; (void)ws_size;

  // ---- inputs (setup_inputs dict order) ----
  const float* X[3] = {(const float*)d_in[0], (const float*)d_in[1], (const float*)d_in[2]};
  const int N[3] = {in_sizes[0] / DIM, in_sizes[1] / DIM, in_sizes[2] / DIM};
  const int*   Lidx[3][3];
  const float* Lval[3][3];
  int nnzL[3];
  {
    int k = 3;
    for (int d = 0; d < 3; ++d) {
      for (int t = 0; t < 3; ++t) {
        Lidx[d][t] = (const int*)d_in[k];
        Lval[d][t] = (const float*)d_in[k + 1];
        k += 2;
      }
      nnzL[d] = in_sizes[4 + d * 6];
    }
  }
  const int*   B1r = (const int*)d_in[21];
  const int*   B1c = (const int*)d_in[22];
  const float* B1v = (const float*)d_in[23];
  const int nnzB1 = in_sizes[23];
  const int*   B2r = (const int*)d_in[24];
  const int*   B2c = (const int*)d_in[25];
  const float* B2v = (const float*)d_in[26];
  const int nnzB2 = in_sizes[26];
  const int* he_idx = (const int*)d_in[27];
  const int* he_seg = (const int*)d_in[28];
  const int HE_TOT_ = in_sizes[27];
  const float* W    = (const float*)d_in[29];
  const float* Wfc  = (const float*)d_in[30];
  const float* Wcls = (const float*)d_in[31];
  const float* bcls = (const float*)d_in[32];
  const int NC_  = in_sizes[31] / DIM;   // 16
  const int NHE_ = 20000;

  // ---- workspace carve ----
  char* ws = (char*)d_ws;
  size_t off = 0;
  auto carve = [&](size_t bytes) -> void* {
    void* p = ws + off;
    off += (bytes + 255) & ~(size_t)255;
    return p;
  };
  // bf16 activation banks: converted input + ping/pong
  unsigned short* xbank[3][3];
  for (int b = 0; b < 3; ++b)
    for (int d = 0; d < 3; ++d)
      xbank[b][d] = (unsigned short*)carve((size_t)N[d] * DIM * sizeof(unsigned short));
  int Nmax = N[0] > N[1] ? N[0] : N[1];
  if (N[2] > Nmax) Nmax = N[2];
  float* T[5];
  unsigned short* Tb[5];
  for (int i = 0; i < 5; ++i) T[i] = (float*)carve((size_t)Nmax * DIM * sizeof(float));
  for (int i = 0; i < 5; ++i)
    Tb[i] = (unsigned short*)carve((size_t)Nmax * DIM * sizeof(unsigned short));
  unsigned short* Wbf   = (unsigned short*)carve((size_t)54 * DIM * DIM * sizeof(unsigned short));
  unsigned short* Whead = (unsigned short*)carve((size_t)DIM * NC_ * sizeof(unsigned short));
  float* pooled = (float*)carve((size_t)NHE_ * DIM * sizeof(float));
  unsigned short* pooledb = (unsigned short*)carve((size_t)NHE_ * DIM * sizeof(unsigned short));
  float* cnt = (float*)carve((size_t)NHE_ * sizeof(float));

  auto gsz = [](long long n, int cap) {
    long long b = (n + 255) / 256;
    if (b > cap) b = cap;
    if (b < 1) b = 1;
    return (int)b;
  };
  auto launch_zero = [&](float* p, size_t nfloats) {
    long long n4 = (long long)(nfloats / 4);
    zero_kernel<<<gsz(n4, 16384), 256, 0, stream>>>((float4*)p, n4);
  };
  auto launch_cvt = [&](const float* src, unsigned short* dst, const float* c, size_t nfloats) {
    long long n4 = (long long)(nfloats / 4);
    cvt_bf16_kernel<<<gsz(n4, 16384), 256, 0, stream>>>(src, dst, c, n4);
  };
  auto launch_spmm = [&](const int* r, const int* c, const float* v,
                         const unsigned short* x, float* out, int nnz) {
    spmm_kernel<<<(nnz + 7) / 8, 256, 0, stream>>>(r, c, v, x, out, nnz);
  };

  // ---- prep: weights to bf16, head fused, inputs to bf16 ----
  prep_w_kernel<<<55, 256, 0, stream>>>(W, Wfc, Wcls, Wbf, Whead, NC_);
  for (int d = 0; d < 3; ++d)
    launch_cvt(X[d], xbank[0][d], nullptr, (size_t)N[d] * DIM);

  // ---- 3 layers ----
  const unsigned short* cur[3] = {xbank[0][0], xbank[0][1], xbank[0][2]};
  for (int l = 0; l < 3; ++l) {
    int bank = 1 + (l & 1);
    unsigned short* nxt[3] = {xbank[bank][0], xbank[bank][1], xbank[bank][2]};
    for (int d = 0; d < 3; ++d) {
      GemmArgs g = {};
      int nterms = 0;
      auto addterm = [&](const unsigned short* a, int term) {
        g.A[nterms] = a;
        g.B[nterms] = Wbf + ((size_t)((l * 3 + d) * 6 + term)) * DIM * DIM;
        ++nterms;
      };
      auto spterm = [&](int slot, const int* r, const int* c, const float* v,
                        const unsigned short* x, int nnz, int term) {
        launch_zero(T[slot], (size_t)N[d] * DIM);
        launch_spmm(r, c, v, x, T[slot], nnz);
        launch_cvt(T[slot], Tb[slot], nullptr, (size_t)N[d] * DIM);
        addterm(Tb[slot], term);
      };
      addterm(cur[d], 0);  // self term
      for (int t = 0; t < 3; ++t)  // L, Lu, Ld
        spterm(t, Lidx[d][t], Lidx[d][t] + nnzL[d], Lval[d][t], cur[d], nnzL[d], t + 1);
      if (d == 0) {
        spterm(3, B1r, B1c, B1v, cur[1], nnzB1, 5);          // from-upper B1 @ x1
      } else if (d == 1) {
        spterm(3, B1c, B1r, B1v, cur[0], nnzB1, 4);          // from-lower B1^T @ x0
        spterm(4, B2r, B2c, B2v, cur[2], nnzB2, 5);          // from-upper B2 @ x2
      } else {
        spterm(3, B2c, B2r, B2v, cur[1], nnzB2, 4);          // from-lower B2^T @ x1
      }
      g.nterms = nterms;
      g.M = N[d];
      g.out = nxt[d];
      g.apply_tanh = 1;
      g.out_bf16 = 1;
      g.bias = nullptr;
      g.drop_first = 0;
      gemm_wmma_kernel<8><<<(N[d] + 255) / 256, 256, 0, stream>>>(g);
    }
    cur[0] = nxt[0]; cur[1] = nxt[1]; cur[2] = nxt[2];
  }

  // ---- hyperedge mean pooling + fused head ----
  launch_zero(pooled, (size_t)NHE_ * DIM);
  launch_zero(cnt, (size_t)NHE_);
  pool_kernel<<<(HE_TOT_ + 7) / 8, 256, 0, stream>>>(he_idx, he_seg, cur[0], pooled, cnt, HE_TOT_);
  launch_cvt(pooled, pooledb, cnt, (size_t)NHE_ * DIM);  // fused mean scale

  GemmArgs h = {};
  h.A[0] = pooledb;
  h.B[0] = Whead;                 // Wfc @ Wcls, bf16 [128][16]
  h.nterms = 1;
  h.M = NHE_;
  h.out = d_out;                  // f32 logits
  h.apply_tanh = 0;
  h.out_bf16 = 0;
  h.bias = bcls;
  h.drop_first = 1;               // logits = p[1:] @ Wcls + b
  gemm_wmma_kernel<1><<<(NHE_ + 255) / 256, 256, 0, stream>>>(h);
}
